// EmbeddingAttention_20925080666218
// MI455X (gfx1250) — matrix operand order repaired and checked by reading, not compiled, not run on hardware
//
#include <hip/hip_runtime.h>
#include <hip/hip_bf16.h>

// CDNA5 / gfx1250 attention layer.
// Matrix math via v_wmma_f32_16x16x32_f16 (f16 inputs, f32 accumulate).
// Flash-attention K/V tiles staged through LDS with the Tensor Data Mover
// (tensor_load_to_lds + s_wait_tensorcnt), double-buffered.

typedef __attribute__((ext_vector_type(16))) _Float16 v16h;
typedef __attribute__((ext_vector_type(8)))  _Float16 v8h;
typedef __attribute__((ext_vector_type(8)))  float    v8f;
typedef __attribute__((ext_vector_type(4)))  unsigned int v4u;
typedef __attribute__((ext_vector_type(8)))  int      v8i;
typedef __attribute__((ext_vector_type(4)))  int      v4i;

#define NHEADS   6
#define NKV      2
#define HDIM     64
#define HIDDEN   384
#define BATCH    4
#define SEQ      2048
#define ROWS     (BATCH * SEQ)            // 8192
#define LN_THETA_OVER_HALFDIM 0.43173468f // ln(1e6)/32

#if __has_builtin(__builtin_amdgcn_tensor_load_to_lds) && \
    __has_builtin(__builtin_amdgcn_s_wait_tensorcnt)
#define HAS_TDM 1
#else
#define HAS_TDM 0
#endif

// ---------------------------------------------------------------------------
// helpers
// ---------------------------------------------------------------------------
__device__ __forceinline__ float red_max16(float v) {
  v = fmaxf(v, __shfl_xor(v, 1, 32));
  v = fmaxf(v, __shfl_xor(v, 2, 32));
  v = fmaxf(v, __shfl_xor(v, 4, 32));
  v = fmaxf(v, __shfl_xor(v, 8, 32));
  return v;
}
__device__ __forceinline__ float red_sum16(float v) {
  v += __shfl_xor(v, 1, 32);
  v += __shfl_xor(v, 2, 32);
  v += __shfl_xor(v, 4, 32);
  v += __shfl_xor(v, 8, 32);
  return v;
}

// A-fragment (16x32 f16, MxK): lane holds row M=lane&15; element e maps to
// k = (lane>=16)*16 + e -> one contiguous 16-half run.
__device__ __forceinline__ v16h load_a_frag(const _Float16* base) {
  v8h g0 = *(const v8h*)(base);
  v8h g1 = *(const v8h*)(base + 8);
  v16h a;
  for (int i = 0; i < 8; ++i) { a[i] = g0[i]; a[i + 8] = g1[i]; }
  return a;
}

#if HAS_TDM
// Issue a 2D TDM load: tile (tile_d1 rows x tile_d0 elems of 2B) from a
// row-major tensor (row stride stride0 elems) at gptr -> LDS offset lds_addr.
// D# layout per CDNA5 ISA ch.8 (group0: count/lds/global/type; group1: dims).
__device__ __forceinline__ void tdm_load_2d(unsigned lds_addr,
                                            const _Float16* gptr,
                                            unsigned tensor_d0, unsigned tensor_d1,
                                            unsigned tile_d0, unsigned tile_d1,
                                            unsigned stride0) {
  unsigned long long ga = (unsigned long long)(size_t)gptr;
  v4u g0;
  g0[0] = 1u;                                   // count=1, user D#, no gather
  g0[1] = lds_addr;                             // LDS byte address
  g0[2] = (unsigned)ga;                         // global_addr[31:0]
  g0[3] = (unsigned)(ga >> 32) | (2u << 30);    // global_addr[56:32] | type=2
  v8i g1;
  g1[0] = (int)0x00010000u;                     // data_size=1 (2 bytes)
  g1[1] = (int)((tensor_d0 & 0xFFFFu) << 16);   // dim0[15:0] (barrier addr = 0)
  g1[2] = (int)((tensor_d0 >> 16) | ((tensor_d1 & 0xFFFFu) << 16));
  g1[3] = (int)((tensor_d1 >> 16) | (tile_d0 << 16));
  g1[4] = (int)tile_d1;                         // tile_dim1 | tile_dim2=0
  g1[5] = (int)stride0;                         // dim0_stride[31:0]
  g1[6] = 0;                                    // dim0_stride hi | dim1_stride lo
  g1[7] = 0;
  v4i z4 = {0, 0, 0, 0};
#if defined(__clang_major__) && __clang_major__ >= 23
  v8i z8 = {0, 0, 0, 0, 0, 0, 0, 0};
  __builtin_amdgcn_tensor_load_to_lds(g0, g1, z4, z4, z8, 0);
#else
  __builtin_amdgcn_tensor_load_to_lds(g0, g1, z4, z4, 0);
#endif
}
#endif

// ---------------------------------------------------------------------------
// Stage 0: f32 -> f16 conversion (grid-stride)
// ---------------------------------------------------------------------------
__global__ void cvt_f32_f16_kernel(const float* __restrict__ src,
                                   _Float16* __restrict__ dst, int n) {
  int i = blockIdx.x * blockDim.x + threadIdx.x;
  int stride = gridDim.x * blockDim.x;
  for (; i < n; i += stride) dst[i] = (_Float16)src[i];
}

// ---------------------------------------------------------------------------
// Stage 1: fused QKV projection + RoPE.
// grid = (ROWS/64, 10); block = 128 (4 waves). Wave: 16 tokens x 64 dims.
// head 0..5 -> Q (+RoPE), 6..7 -> K (+RoPE), 8..9 -> V (stored transposed).
// ---------------------------------------------------------------------------
__global__ __launch_bounds__(128)
void qkv_rope_kernel(const _Float16* __restrict__ h16,
                     const _Float16* __restrict__ wq16,
                     const _Float16* __restrict__ wk16,
                     const _Float16* __restrict__ wv16,
                     _Float16* __restrict__ qb,   // [B][6][S][64]
                     _Float16* __restrict__ kb,   // [B][2][S][64]
                     _Float16* __restrict__ vtb)  // [B][2][64][S]
{
  const int wave = threadIdx.x >> 5;
  const int lane = threadIdx.x & 31;
  const int half = lane >> 4;
  const int c    = lane & 15;
  const int row0 = (blockIdx.x * 4 + wave) * 16;
  const int head = blockIdx.y;

  const _Float16* w;
  int wrow0;
  if (head < 6)      { w = wq16; wrow0 = head * 64; }
  else if (head < 8) { w = wk16; wrow0 = (head - 6) * 64; }
  else               { w = wv16; wrow0 = (head - 8) * 64; }

  v8f acc[4];
  for (int j = 0; j < 4; ++j)
    for (int r = 0; r < 8; ++r) acc[j][r] = 0.0f;

  for (int k0 = 0; k0 < HIDDEN; k0 += 32) {
    const _Float16* abase = h16 + (size_t)(row0 + c) * HIDDEN + k0 + half * 16;
    if (k0 + 32 < HIDDEN)
      __builtin_prefetch(abase + 32, 0, 0);  // gfx1250 global_prefetch_b8
    v16h a = load_a_frag(abase);
    for (int j = 0; j < 4; ++j) {
      v16h b = *(const v16h*)(w + (size_t)(wrow0 + 16 * j + c) * HIDDEN + k0 + 16 * half);
      acc[j] = __builtin_amdgcn_wmma_f32_16x16x32_f16(
          false, a, false, b, (short)0, acc[j], false, false);
    }
  }

  if (head < 8) {
    // RoPE pairs (d, d+32): d = 16*j + c in acc[j], partner in acc[j+2]
    for (int j = 0; j < 2; ++j) {
      const float inv = __expf(-(float)(16 * j + c) * LN_THETA_OVER_HALFDIM);
      for (int r = 0; r < 8; ++r) {
        const int s = (row0 + r + half * 8) & (SEQ - 1);
        float cs, sn;
        __sincosf((float)s * inv, &sn, &cs);
        const float lo = acc[j][r], hi = acc[j + 2][r];
        acc[j][r]     = lo * cs - hi * sn;
        acc[j + 2][r] = hi * cs + lo * sn;
      }
    }
  }

  for (int j = 0; j < 4; ++j) {
    for (int r = 0; r < 8; ++r) {
      const int row = row0 + r + half * 8;
      const int b   = row >> 11;          // / SEQ
      const int s   = row & (SEQ - 1);
      const int d   = 16 * j + c;
      const _Float16 val = (_Float16)acc[j][r];
      if (head < 6)
        qb[(((size_t)b * 6 + head) * SEQ + s) * 64 + d] = val;
      else if (head < 8)
        kb[(((size_t)b * 2 + (head - 6)) * SEQ + s) * 64 + d] = val;
      else
        vtb[(((size_t)b * 2 + (head - 8)) * 64 + d) * SEQ + s] = val;
    }
  }
}

// ---------------------------------------------------------------------------
// Stage 2: flash attention. grid = (SEQ/16, 6, BATCH); block = 32 (one wave).
// 32 keys/iter: 4 score WMMAs + online softmax + P via LDS relayout + 4 PV
// WMMAs. K/V tiles DMA'd into double-buffered LDS by the TDM when available.
// ---------------------------------------------------------------------------
__global__ __launch_bounds__(32)
void flash_attn_kernel(const _Float16* __restrict__ qb,
                       const _Float16* __restrict__ kb,
                       const _Float16* __restrict__ vtb,
                       const int* __restrict__ amask,
                       _Float16* __restrict__ attn)  // [B*S][384]
{
#if HAS_TDM
  __shared__ __align__(32) _Float16 kls[2][32][64];  // K tiles: 2 x 4KB
  __shared__ __align__(32) _Float16 vls[2][64][32];  // V tiles: 2 x 4KB
#endif
  __shared__ __align__(32) _Float16 pls[16 * 32];    // P tile (C->A relayout)

  const int lane = threadIdx.x & 31;
  const int half = lane >> 4;
  const int c    = lane & 15;
  const int q0   = blockIdx.x * 16;
  const int h    = blockIdx.y;
  const int b    = blockIdx.z;
  const int kv   = h / (NHEADS / NKV);   // GQA: 3 query heads per kv head

  const _Float16* qptr = qb  + (((size_t)b * 6 + h)  * SEQ) * 64;
  const _Float16* kptr = kb  + (((size_t)b * 2 + kv) * SEQ) * 64;
  const _Float16* vptr = vtb + (((size_t)b * 2 + kv) * 64) * SEQ;
  const int*      mrow = amask + (size_t)b * SEQ;

  // Q A-fragments for both 32-wide d-chunks
  v16h aq[2];
  for (int ch = 0; ch < 2; ++ch)
    aq[ch] = load_a_frag(qptr + (size_t)(q0 + c) * 64 + 32 * ch + half * 16);

  float mrun[8], lrun[8];
  v8f accv[4];
  for (int r = 0; r < 8; ++r) { mrun[r] = -3.0e38f; lrun[r] = 0.0f; }
  for (int j = 0; j < 4; ++j)
    for (int r = 0; r < 8; ++r) accv[j][r] = 0.0f;

  const int NT = SEQ / 32;

#if HAS_TDM
  // Prologue: DMA tile 0 into buffer 0 (2 tensor ops in flight).
  tdm_load_2d((unsigned)(size_t)&kls[0][0][0], kptr,            64, SEQ, 64, 32, 64);
  tdm_load_2d((unsigned)(size_t)&vls[0][0][0], vptr,            SEQ, 64, 32, 64, SEQ);
#endif

  for (int kt = 0; kt < NT; ++kt) {
    const int key0 = kt * 32;
#if HAS_TDM
    const int cur = kt & 1;
    if (kt + 1 < NT) {
      // WAR guard: LDS reads of buffer (kt+1)&1 from iteration kt-1 must
      // retire before the TDM overwrites it.
      asm volatile("s_wait_dscnt 0x0" ::: "memory");
      const int key1 = key0 + 32;
      tdm_load_2d((unsigned)(size_t)&kls[cur ^ 1][0][0], kptr + (size_t)key1 * 64,
                  64, SEQ, 64, 32, 64);
      tdm_load_2d((unsigned)(size_t)&vls[cur ^ 1][0][0], vptr + key1,
                  SEQ, 64, 32, 64, SEQ);
      __builtin_amdgcn_s_wait_tensorcnt((short)2);  // current buffer ready
    } else {
      __builtin_amdgcn_s_wait_tensorcnt((short)0);
    }
#endif

    // scores: S = Q K^T  (two 16-key column tiles, two d-chunks each)
    v8f sc[2];
    for (int nt = 0; nt < 2; ++nt)
      for (int r = 0; r < 8; ++r) sc[nt][r] = 0.0f;
    for (int nt = 0; nt < 2; ++nt) {
      for (int ch = 0; ch < 2; ++ch) {
#if HAS_TDM
        v16h bf = *(const v16h*)(&kls[cur][nt * 16 + c][32 * ch + 16 * half]);
#else
        v16h bf = *(const v16h*)(kptr + (size_t)(key0 + nt * 16 + c) * 64 + 32 * ch + 16 * half);
#endif
        sc[nt] = __builtin_amdgcn_wmma_f32_16x16x32_f16(
            false, aq[ch], false, bf, (short)0, sc[nt], false, false);
      }
    }

    // scale 1/sqrt(64) + attention mask (per key column, per lane)
    float madd[2];
    for (int nt = 0; nt < 2; ++nt)
      madd[nt] = (mrow[key0 + nt * 16 + c] == 0) ? -1.0e30f : 0.0f;
    for (int nt = 0; nt < 2; ++nt)
      for (int r = 0; r < 8; ++r)
        sc[nt][r] = sc[nt][r] * 0.125f + madd[nt];

    // online softmax, per query row r (row m = r + half*8)
    for (int r = 0; r < 8; ++r) {
      float t  = red_max16(fmaxf(sc[0][r], sc[1][r]));
      float nm = fmaxf(mrun[r], t);
      float sf = __expf(mrun[r] - nm);
      mrun[r] = nm;
      lrun[r] *= sf;
      for (int j = 0; j < 4; ++j) accv[j][r] *= sf;
      float p0 = __expf(sc[0][r] - nm);
      float p1 = __expf(sc[1][r] - nm);
      sc[0][r] = p0; sc[1][r] = p1;
      lrun[r] += red_sum16(p0 + p1);
    }

    // P: C-layout -> LDS -> A-layout for the PV WMMA
    for (int nt = 0; nt < 2; ++nt)
      for (int r = 0; r < 8; ++r)
        pls[(r + half * 8) * 32 + nt * 16 + c] = (_Float16)sc[nt][r];
    __syncthreads();
    v16h ap = load_a_frag(pls + c * 32 + half * 16);
    __syncthreads();

    // O += P V   (V transposed [D][S] -> contiguous B-fragment loads)
    for (int j = 0; j < 4; ++j) {
#if HAS_TDM
      v16h vf = *(const v16h*)(&vls[cur][16 * j + c][16 * half]);
#else
      v16h vf = *(const v16h*)(vptr + (size_t)(16 * j + c) * SEQ + key0 + 16 * half);
#endif
      accv[j] = __builtin_amdgcn_wmma_f32_16x16x32_f16(
          false, ap, false, vf, (short)0, accv[j], false, false);
    }
  }

  // normalize and emit to [B*S][H*64] f16
  for (int j = 0; j < 4; ++j)
    for (int r = 0; r < 8; ++r) {
      const float o = accv[j][r] / lrun[r];
      attn[(size_t)(b * SEQ + q0 + r + half * 8) * HIDDEN + h * 64 + 16 * j + c] = (_Float16)o;
    }
}

// ---------------------------------------------------------------------------
// Stage 3: output projection attn @ wo^T -> f32. grid = (ROWS/64, 6), block 128.
// ---------------------------------------------------------------------------
__global__ __launch_bounds__(128)
void outproj_kernel(const _Float16* __restrict__ a16,
                    const _Float16* __restrict__ wo16,
                    float* __restrict__ out) {
  const int wave = threadIdx.x >> 5;
  const int lane = threadIdx.x & 31;
  const int half = lane >> 4;
  const int c    = lane & 15;
  const int row0 = (blockIdx.x * 4 + wave) * 16;
  const int strip = blockIdx.y;

  v8f acc[4];
  for (int j = 0; j < 4; ++j)
    for (int r = 0; r < 8; ++r) acc[j][r] = 0.0f;

  for (int k0 = 0; k0 < HIDDEN; k0 += 32) {
    const _Float16* abase = a16 + (size_t)(row0 + c) * HIDDEN + k0 + half * 16;
    if (k0 + 32 < HIDDEN)
      __builtin_prefetch(abase + 32, 0, 0);
    v16h a = load_a_frag(abase);
    for (int j = 0; j < 4; ++j) {
      v16h b = *(const v16h*)(wo16 + (size_t)(strip * 64 + 16 * j + c) * HIDDEN + k0 + 16 * half);
      acc[j] = __builtin_amdgcn_wmma_f32_16x16x32_f16(
          false, a, false, b, (short)0, acc[j], false, false);
    }
  }

  for (int j = 0; j < 4; ++j)
    for (int r = 0; r < 8; ++r)
      out[(size_t)(row0 + r + half * 8) * HIDDEN + strip * 64 + 16 * j + c] = acc[j][r];
}

// ---------------------------------------------------------------------------
// launcher
// ---------------------------------------------------------------------------
extern "C" void kernel_launch(void* const* d_in, const int* in_sizes, int n_in,
                              void* d_out, int out_size, void* d_ws, size_t ws_size,
                              hipStream_t stream) {
  (void)in_sizes; (void)n_in; (void)out_size; (void)ws_size;

  const float* hs   = (const float*)d_in[0];
  const int*   mask = (const int*)d_in[1];
  const float* wq   = (const float*)d_in[2];
  const float* wk   = (const float*)d_in[3];
  const float* wv   = (const float*)d_in[4];
  const float* wo   = (const float*)d_in[5];
  float* out = (float*)d_out;

  // carve workspace (256B aligned), ~24 MB total
  char* ws = (char*)d_ws;
  size_t off = 0;
  auto alloc = [&](size_t bytes) -> void* {
    void* p = ws + off;
    off = (off + bytes + 255) & ~(size_t)255;
    return p;
  };
  _Float16* h16   = (_Float16*)alloc((size_t)ROWS * HIDDEN * 2);
  _Float16* wq16  = (_Float16*)alloc((size_t)HIDDEN * HIDDEN * 2);
  _Float16* wk16  = (_Float16*)alloc((size_t)NKV * HDIM * HIDDEN * 2);
  _Float16* wv16  = (_Float16*)alloc((size_t)NKV * HDIM * HIDDEN * 2);
  _Float16* wo16  = (_Float16*)alloc((size_t)HIDDEN * HIDDEN * 2);
  _Float16* qb    = (_Float16*)alloc((size_t)BATCH * NHEADS * SEQ * HDIM * 2);
  _Float16* kb    = (_Float16*)alloc((size_t)BATCH * NKV * SEQ * HDIM * 2);
  _Float16* vtb   = (_Float16*)alloc((size_t)BATCH * NKV * HDIM * SEQ * 2);
  _Float16* at16  = (_Float16*)alloc((size_t)ROWS * HIDDEN * 2);

  cvt_f32_f16_kernel<<<512, 256, 0, stream>>>(hs, h16, ROWS * HIDDEN);
  cvt_f32_f16_kernel<<<256, 256, 0, stream>>>(wq, wq16, HIDDEN * HIDDEN);
  cvt_f32_f16_kernel<<<128, 256, 0, stream>>>(wk, wk16, NKV * HDIM * HIDDEN);
  cvt_f32_f16_kernel<<<128, 256, 0, stream>>>(wv, wv16, NKV * HDIM * HIDDEN);
  cvt_f32_f16_kernel<<<256, 256, 0, stream>>>(wo, wo16, HIDDEN * HIDDEN);

  qkv_rope_kernel<<<dim3(ROWS / 64, 10), 128, 0, stream>>>(
      h16, wq16, wk16, wv16, qb, kb, vtb);

  flash_attn_kernel<<<dim3(SEQ / 16, NHEADS, BATCH), 32, 0, stream>>>(
      qb, kb, vtb, mask, at16);

  outproj_kernel<<<dim3(ROWS / 64, NHEADS), 128, 0, stream>>>(at16, wo16, out);
}
